// GRKULayer_72584947302763
// MI455X (gfx1250) — matrix operand order repaired and checked by reading, not compiled, not run on hardware
//
#include <hip/hip_runtime.h>
#include <hip/hip_bf16.h>

// ---------------------------------------------------------------------------
// GRKU (GRU-KAN) forward for MI455X (gfx1250), wave32 + WMMA bf16 path.
// All GEMMs run through v_wmma_f32_16x16x32_bf16 (fp32 accumulate).
// Each wave owns a 16x32 output patch: one A fragment feeds two independent
// WMMAs, hiding the bf16 WMMA WAR-hazard NOP slots with the extra B loads.
// ---------------------------------------------------------------------------

#define BATCH 256
#define TSEQ  128
#define INPUT 256
#define HID   512
#define KANH  64
#define LN_EPS 1e-5f

typedef __attribute__((ext_vector_type(16))) __bf16 v16bf;
typedef __attribute__((ext_vector_type(8)))  float  v8f;

union BV { unsigned int u[8]; v16bf v; };

__device__ __forceinline__ unsigned short f32_to_bf16_bits(float f) {
    unsigned int u = __float_as_uint(f);
    u += 0x7FFFu + ((u >> 16) & 1u);           // round-to-nearest-even
    return (unsigned short)(u >> 16);
}
__device__ __forceinline__ float silu_f(float x) {
    return x / (1.0f + __expf(-x));
}

// ---------------------------------------------------------------------------
// Generic GEMM: C(MxN) = A(MxK) @ W(NxK)^T [+ bias] [+= C]
// A is fp32 (converted on load) or raw-bf16; W is raw-bf16 (preconverted).
// One 16x32 output patch per wave (two 16x16 WMMA tiles sharing the A frag);
// K-loop in steps of 32 with ISA-documented per-lane VGPR layouts.
// Requires N % 32 == 0, M % 16 == 0, K % 32 == 0 (all call sites satisfy).
// ---------------------------------------------------------------------------
template <bool AF32>
__global__ __launch_bounds__(256)
void gemm_wmma_bf16(const void* __restrict__ Araw, int lda,
                    const unsigned short* __restrict__ W,   // N x K bf16 bits
                    const float* __restrict__ bias,
                    float* __restrict__ C, int ldc,
                    int M, int N, int K, int accumulate)
{
    const int wave = threadIdx.x >> 5;
    const int lane = threadIdx.x & 31;
    const int half = lane >> 4;        // 0: lanes 0-15, 1: lanes 16-31
    const int rm   = lane & 15;

    const int tilesN = N >> 5;                     // 32-wide N patches
    const int tile   = blockIdx.x * 8 + wave;
    if (tile >= (M >> 4) * tilesN) return;         // wave-uniform exit
    const int tm = tile / tilesN;
    const int tn = tile - tm * tilesN;

    const int arow  = tm * 16 + rm;    // A row held by this lane (both halves)
    const int wrow0 = tn * 32 + rm;    // W row == output column, tile 0
    const int wrow1 = wrow0 + 16;      // W row == output column, tile 1

    const float*          Af  = (const float*)Araw;
    const unsigned short* Ab  = (const unsigned short*)Araw;
    const unsigned short* Wr0 = W + (size_t)wrow0 * K;
    const unsigned short* Wr1 = W + (size_t)wrow1 * K;

    v8f acc0 = {0.f,0.f,0.f,0.f,0.f,0.f,0.f,0.f};
    v8f acc1 = {0.f,0.f,0.f,0.f,0.f,0.f,0.f,0.f};

    for (int k0 = 0; k0 < K; k0 += 32) {
        BV a, b0, b1;
#pragma unroll
        for (int j = 0; j < 8; ++j) {
            // A 16x32 bf16 layout: VGPR j holds K = (j&3)*2 + half*8 + (j>>2)*16
            const int ka = k0 + ((j & 3) << 1) + (half << 3) + ((j >> 2) << 4);
            if (AF32) {
                const float f0 = Af[(size_t)arow * lda + ka];
                const float f1 = Af[(size_t)arow * lda + ka + 1];
                a.u[j] = (unsigned)f32_to_bf16_bits(f0)
                       | ((unsigned)f32_to_bf16_bits(f1) << 16);
            } else {
                a.u[j] = *(const unsigned int*)(Ab + (size_t)arow * lda + ka);
            }
            // B 32x16 bf16 layout: VGPR j holds K = (j&3)*2 + half*8 + (j>>2)*16 (cols in lanes)
            const int kb = k0 + ((j & 3) << 1) + (half << 3) + ((j >> 2) << 4);
            b0.u[j] = *(const unsigned int*)(Wr0 + kb);
            b1.u[j] = *(const unsigned int*)(Wr1 + kb);
        }
        if (k0 + 32 < K) {                 // speculative L2 prefetch of next K-block
            __builtin_prefetch((const void*)(Wr0 + k0 + 32), 0, 1);
            __builtin_prefetch((const void*)(Wr1 + k0 + 32), 0, 1);
            if (!AF32) __builtin_prefetch((const void*)(Ab + (size_t)arow * lda + k0 + 32), 0, 1);
        }
        acc0 = __builtin_amdgcn_wmma_f32_16x16x32_bf16(
                   false, a.v, false, b0.v, (short)0, acc0, false, false);
        acc1 = __builtin_amdgcn_wmma_f32_16x16x32_bf16(
                   false, a.v, false, b1.v, (short)0, acc1, false, false);
    }

    const float bv0 = bias ? bias[wrow0] : 0.0f;
    const float bv1 = bias ? bias[wrow1] : 0.0f;
#pragma unroll
    for (int r = 0; r < 8; ++r) {
        // C/D layout: VGPR r -> M = r + 8*half, N = lane%16 (per tile)
        const int m = tm * 16 + r + (half << 3);
        float* cp0 = C + (size_t)m * ldc + wrow0;
        float* cp1 = C + (size_t)m * ldc + wrow1;
        float v0 = acc0[r] + bv0;
        float v1 = acc1[r] + bv1;
        if (accumulate) { v0 += *cp0; v1 += *cp1; }
        *cp0 = v0;
        *cp1 = v1;
    }
}

// ---------------------------------------------------------------------------
// Cox-de-Boor order-3 B-spline bases on a uniform grid in [-1,1].
// ---------------------------------------------------------------------------
template <int GSZ>
__device__ __forceinline__ void bsplines3(float x, float* __restrict__ out /* GSZ+3 */)
{
    const float hg = 2.0f / (float)GSZ;
    float bb[GSZ + 6];
#pragma unroll
    for (int i = 0; i < GSZ + 6; ++i) {
        const float t0 = -1.0f + (float)(i - 3) * hg;
        bb[i] = (x >= t0 && x < t0 + hg) ? 1.0f : 0.0f;
    }
#pragma unroll
    for (int p = 1; p <= 3; ++p) {
        const float inv = 1.0f / ((float)p * hg);
#pragma unroll
        for (int i = 0; i < GSZ + 6 - p; ++i) {
            const float ti   = -1.0f + (float)(i - 3) * hg;      // t[i]
            const float tip1 = -1.0f + (float)(i + p - 2) * hg;  // t[i+p+1]
            bb[i] = (x - ti) * inv * bb[i] + (tip1 - x) * inv * bb[i + 1];
        }
    }
#pragma unroll
    for (int c = 0; c < GSZ + 3; ++c) out[c] = bb[c];
}

// comb = [x_proj, h_prev]; emit bf16 comb, bf16 silu(comb), bf16 spline bases.
template <int GSZ>
__global__ __launch_bounds__(256)
void prep_comb(const float* __restrict__ xproj, const float* __restrict__ h,
               unsigned short* __restrict__ comb_bf,
               unsigned short* __restrict__ silu_bf,
               unsigned short* __restrict__ bases_bf)
{
    const int NC = GSZ + 3;
    const int idx = blockIdx.x * blockDim.x + threadIdx.x;
    if (idx >= BATCH * 2 * HID) return;
    const int b = idx >> 10;            // / 1024
    const int j = idx & 1023;
    const float x = (j < HID) ? xproj[(size_t)b * HID + j]
                              : h[(size_t)b * HID + (j - HID)];
    comb_bf[idx] = f32_to_bf16_bits(x);
    silu_bf[idx] = f32_to_bf16_bits(silu_f(x));
    float bs[GSZ + 3];
    bsplines3<GSZ>(x, bs);
    unsigned short* o = bases_bf + (size_t)idx * NC;
#pragma unroll
    for (int c = 0; c < NC; ++c) o[c] = f32_to_bf16_bits(bs[c]);
}

// h1 -> bf16 silu(h1), bf16 spline bases (KAN hidden width 64).
template <int GSZ>
__global__ __launch_bounds__(256)
void prep_h1(const float* __restrict__ h1,
             unsigned short* __restrict__ silu_bf,
             unsigned short* __restrict__ bases_bf)
{
    const int NC = GSZ + 3;
    const int idx = blockIdx.x * blockDim.x + threadIdx.x;
    if (idx >= BATCH * KANH) return;
    const float x = h1[idx];
    silu_bf[idx] = f32_to_bf16_bits(silu_f(x));
    float bs[GSZ + 3];
    bsplines3<GSZ>(x, bs);
    unsigned short* o = bases_bf + (size_t)idx * NC;
#pragma unroll
    for (int c = 0; c < NC; ++c) o[c] = f32_to_bf16_bits(bs[c]);
}

// ---------------------------------------------------------------------------
// Fused gate: update=sigmoid(LN(u_pre)); hc=silu(LN(LN(cand+resid)));
// h_cell=(1-u)*h_prev+u*hc; h_new=LN(h_cell+h_prev). One block per batch row.
// ---------------------------------------------------------------------------
__global__ __launch_bounds__(256)
void gate_fuse(const float* __restrict__ u_pre, const float* __restrict__ cand,
               const float* __restrict__ resid, float* __restrict__ h,
               const float* __restrict__ lug, const float* __restrict__ lub,
               const float* __restrict__ cg,  const float* __restrict__ cb,
               const float* __restrict__ ng,  const float* __restrict__ nb,
               float* __restrict__ out_slot, float* __restrict__ last_slot)
{
    __shared__ float sred[256];
    const int b = blockIdx.x, tid = threadIdx.x;
    const int j0 = tid, j1 = tid + 256;
    const size_t base = (size_t)b * HID;

    auto reduceSum = [&](float v) -> float {
        sred[tid] = v; __syncthreads();
        for (int s = 128; s > 0; s >>= 1) {
            if (tid < s) sred[tid] += sred[tid + s];
            __syncthreads();
        }
        const float r = sred[0]; __syncthreads();
        return r;
    };
    auto ln2 = [&](float& v0, float& v1, const float* g, const float* be) {
        const float m  = reduceSum(v0 + v1) * (1.0f / 512.0f);
        const float d0 = v0 - m, d1 = v1 - m;
        const float vv = reduceSum(d0 * d0 + d1 * d1) * (1.0f / 512.0f);
        const float r  = rsqrtf(vv + LN_EPS);
        v0 = d0 * r * g[j0] + be[j0];
        v1 = d1 * r * g[j1] + be[j1];
    };

    float u0 = u_pre[base + j0], u1 = u_pre[base + j1];
    ln2(u0, u1, lug, lub);
    u0 = 1.0f / (1.0f + __expf(-u0));
    u1 = 1.0f / (1.0f + __expf(-u1));

    float c0 = cand[base + j0] + resid[base + j0];
    float c1 = cand[base + j1] + resid[base + j1];
    ln2(c0, c1, cg, cb);
    ln2(c0, c1, cg, cb);            // candidate_norm applied twice (as in source)
    c0 = silu_f(c0); c1 = silu_f(c1);

    const float hp0 = h[base + j0], hp1 = h[base + j1];
    float v0 = ((1.0f - u0) * hp0 + u0 * c0) + hp0;
    float v1 = ((1.0f - u1) * hp1 + u1 * c1) + hp1;
    ln2(v0, v1, ng, nb);

    h[base + j0] = v0; h[base + j1] = v1;
    if (out_slot)  { out_slot[(size_t)b * (2 * HID) + j0] = v0;
                     out_slot[(size_t)b * (2 * HID) + j1] = v1; }
    if (last_slot) { last_slot[base + j0] = v0; last_slot[base + j1] = v1; }
}

__global__ void cvt_f32_bf16(const float* __restrict__ s,
                             unsigned short* __restrict__ d, int n)
{
    const int i = blockIdx.x * blockDim.x + threadIdx.x;
    if (i < n) d[i] = f32_to_bf16_bits(s[i]);
}

__global__ void zero_f32(float* __restrict__ p, int n)
{
    const int i = blockIdx.x * blockDim.x + threadIdx.x;
    if (i < n) p[i] = 0.0f;
}

// ---------------------------------------------------------------------------
// Host side
// ---------------------------------------------------------------------------
static void launch_gemm(bool af32, const void* A, int lda,
                        const unsigned short* W, const float* bias,
                        float* C, int ldc, int M, int N, int K, int acc,
                        hipStream_t s)
{
    const int tiles = (M / 16) * (N / 32);   // 16x32 patch per wave
    dim3 g((tiles + 7) / 8), blk(256);
    if (af32) gemm_wmma_bf16<true ><<<g, blk, 0, s>>>(A, lda, W, bias, C, ldc, M, N, K, acc);
    else      gemm_wmma_bf16<false><<<g, blk, 0, s>>>(A, lda, W, bias, C, ldc, M, N, K, acc);
}

extern "C" void kernel_launch(void* const* d_in, const int* in_sizes, int n_in,
                              void* d_out, int out_size, void* d_ws, size_t ws_size,
                              hipStream_t stream)
{
    (void)in_sizes; (void)n_in; (void)out_size; (void)ws_size;
    const float* x = (const float*)d_in[0];
    float* out = (float*)d_out;

    // input dict order: x, then per cell {Wi,bi,Wr,br,lnr_g,lnr_b,Wu,bu,lnu_g,
    // lnu_b,kan0_base,kan0_spline,kan1_base,kan1_spline,cand_g,cand_b,Wres,bres},
    // then norms (g,b) x2.  Reset branch (Wr/br/lnr*) is dead code -> skipped.
    struct Layer {
        const float *Wi,*bi,*Wu,*bu,*lug,*lub,*k0b,*k0s,*k1b,*k1s,*cg,*cb,*Wres,*bres,*ng,*nb;
        int in_dim, g, nc;
        unsigned short *Wi_bf,*Wu_bf,*Wres_bf,*k0b_bf,*k0s_bf,*k1b_bf,*k1s_bf;
    } L[2];
    for (int i = 0; i < 2; ++i) {
        const int c = 1 + 18 * i;
        L[i].Wi  = (const float*)d_in[c + 0];  L[i].bi   = (const float*)d_in[c + 1];
        L[i].Wu  = (const float*)d_in[c + 6];  L[i].bu   = (const float*)d_in[c + 7];
        L[i].lug = (const float*)d_in[c + 8];  L[i].lub  = (const float*)d_in[c + 9];
        L[i].k0b = (const float*)d_in[c + 10]; L[i].k0s  = (const float*)d_in[c + 11];
        L[i].k1b = (const float*)d_in[c + 12]; L[i].k1s  = (const float*)d_in[c + 13];
        L[i].cg  = (const float*)d_in[c + 14]; L[i].cb   = (const float*)d_in[c + 15];
        L[i].Wres= (const float*)d_in[c + 16]; L[i].bres = (const float*)d_in[c + 17];
        L[i].ng  = (const float*)d_in[37 + 2 * i];
        L[i].nb  = (const float*)d_in[38 + 2 * i];
        L[i].in_dim = (i == 0) ? INPUT : HID;
        L[i].g  = 5 + i;
        L[i].nc = L[i].g + 3;
    }

    // ---- workspace carve-out (256B aligned) ----
    char* wp = (char*)d_ws;
    auto carve = [&](size_t bytes) -> void* {
        void* p = (void*)wp;
        wp += (bytes + 255) & ~(size_t)255;
        return p;
    };
    float* xproj = (float*)carve((size_t)BATCH * HID * 4);
    float* u_pre = (float*)carve((size_t)BATCH * HID * 4);
    float* resid = (float*)carve((size_t)BATCH * HID * 4);
    float* cand  = (float*)carve((size_t)BATCH * HID * 4);
    float* h1    = (float*)carve((size_t)BATCH * KANH * 4);
    float* h     = (float*)carve((size_t)BATCH * HID * 4);
    unsigned short* comb_bf   = (unsigned short*)carve((size_t)BATCH * 2 * HID * 2);
    unsigned short* siluc_bf  = (unsigned short*)carve((size_t)BATCH * 2 * HID * 2);
    unsigned short* bases0_bf = (unsigned short*)carve((size_t)BATCH * 2 * HID * 9 * 2);
    unsigned short* siluh_bf  = (unsigned short*)carve((size_t)BATCH * KANH * 2);
    unsigned short* bases1_bf = (unsigned short*)carve((size_t)BATCH * KANH * 9 * 2);
    for (int i = 0; i < 2; ++i) {
        L[i].Wi_bf   = (unsigned short*)carve((size_t)HID * L[i].in_dim * 2);
        L[i].Wu_bf   = (unsigned short*)carve((size_t)HID * 2 * HID * 2);
        L[i].Wres_bf = (unsigned short*)carve((size_t)HID * 2 * HID * 2);
        L[i].k0b_bf  = (unsigned short*)carve((size_t)KANH * 2 * HID * 2);
        L[i].k0s_bf  = (unsigned short*)carve((size_t)KANH * 2 * HID * L[i].nc * 2);
        L[i].k1b_bf  = (unsigned short*)carve((size_t)HID * KANH * 2);
        L[i].k1s_bf  = (unsigned short*)carve((size_t)HID * KANH * L[i].nc * 2);
    }

    // ---- weight fp32 -> bf16 (each call; deterministic) ----
    auto cvt = [&](const float* s, unsigned short* d, int n) {
        cvt_f32_bf16<<<(n + 255) / 256, 256, 0, stream>>>(s, d, n);
    };
    for (int i = 0; i < 2; ++i) {
        cvt(L[i].Wi,   L[i].Wi_bf,   HID * L[i].in_dim);
        cvt(L[i].Wu,   L[i].Wu_bf,   HID * 2 * HID);
        cvt(L[i].Wres, L[i].Wres_bf, HID * 2 * HID);
        cvt(L[i].k0b,  L[i].k0b_bf,  KANH * 2 * HID);
        cvt(L[i].k0s,  L[i].k0s_bf,  KANH * 2 * HID * L[i].nc);
        cvt(L[i].k1b,  L[i].k1b_bf,  HID * KANH);
        cvt(L[i].k1s,  L[i].k1s_bf,  HID * KANH * L[i].nc);
    }
    zero_f32<<<(BATCH * HID + 255) / 256, 256, 0, stream>>>(h, BATCH * HID);

    // ---- recurrence ----
    for (int t = 0; t < TSEQ; ++t) {
        for (int l = 0; l < 2; ++l) {
            const Layer& c = L[l];
            const void* Ax = (l == 0) ? (const void*)(x + (size_t)t * INPUT)
                                      : (const void*)h;
            const int lda = (l == 0) ? TSEQ * INPUT : HID;

            // x_proj = x_t @ Wi^T + bi
            launch_gemm(true, Ax, lda, c.Wi_bf, c.bi, xproj, HID,
                        BATCH, HID, c.in_dim, 0, stream);

            // comb / silu(comb) / spline bases (bf16 operands)
            const int nP = BATCH * 2 * HID;
            if (l == 0) prep_comb<5><<<(nP + 255) / 256, 256, 0, stream>>>(
                            xproj, h, comb_bf, siluc_bf, bases0_bf);
            else        prep_comb<6><<<(nP + 255) / 256, 256, 0, stream>>>(
                            xproj, h, comb_bf, siluc_bf, bases0_bf);

            // update & residual gates
            launch_gemm(false, comb_bf, 2 * HID, c.Wu_bf,   c.bu,   u_pre, HID,
                        BATCH, HID, 2 * HID, 0, stream);
            launch_gemm(false, comb_bf, 2 * HID, c.Wres_bf, c.bres, resid, HID,
                        BATCH, HID, 2 * HID, 0, stream);

            // KAN layer 0: h1 = silu(comb)@base^T + bases@spline^T
            launch_gemm(false, siluc_bf, 2 * HID, c.k0b_bf, nullptr, h1, KANH,
                        BATCH, KANH, 2 * HID, 0, stream);
            launch_gemm(false, bases0_bf, 2 * HID * c.nc, c.k0s_bf, nullptr, h1, KANH,
                        BATCH, KANH, 2 * HID * c.nc, 1, stream);

            const int nQ = BATCH * KANH;
            if (l == 0) prep_h1<5><<<(nQ + 255) / 256, 256, 0, stream>>>(
                            h1, siluh_bf, bases1_bf);
            else        prep_h1<6><<<(nQ + 255) / 256, 256, 0, stream>>>(
                            h1, siluh_bf, bases1_bf);

            // KAN layer 1: cand
            launch_gemm(false, siluh_bf, KANH, c.k1b_bf, nullptr, cand, HID,
                        BATCH, HID, KANH, 0, stream);
            launch_gemm(false, bases1_bf, KANH * c.nc, c.k1s_bf, nullptr, cand, HID,
                        BATCH, HID, KANH * c.nc, 1, stream);

            float* out_slot  = (t == TSEQ - 1) ? out + (size_t)l * HID : nullptr;
            float* last_slot = (t == TSEQ - 1 && l == 1)
                                   ? out + (size_t)BATCH * 2 * HID : nullptr;
            gate_fuse<<<BATCH, 256, 0, stream>>>(u_pre, cand, resid, h,
                                                 c.lug, c.lub, c.cg, c.cb,
                                                 c.ng, c.nb, out_slot, last_slot);
        }
    }
}
